// ShiftLocalAttention2d_9749575762382
// MI455X (gfx1250) — matrix operand order repaired and checked by reading, not compiled, not run on hardware
//
#include <hip/hip_runtime.h>

typedef __attribute__((ext_vector_type(16))) _Float16 v16h;
typedef __attribute__((ext_vector_type(8)))  float    v8f;

#define BATCH   4
#define HEADS   8
#define DHEAD   32
#define IMH     48
#define IMW     48
#define RAD     3          // window 7 -> radius 3
#define TQ      4          // 4x4 query tile = 16 queries (WMMA M)
#define SLAB    10         // TQ + 2*RAD
#define SLAB2   100        // 10*10 slab positions
#define NT      7          // ceil(100/16) score n-tiles
#define SLABP   112        // NT*16  (K slab rows, padded)
#define SLABPV  128        // padded to 4 K-steps of 32 for P*V

#if __has_builtin(__builtin_amdgcn_global_load_async_to_lds_b32)
#define HAVE_ASYNC_LDS 1
#else
#define HAVE_ASYNC_LDS 0
#endif

// the async builtins take generic int* for both global src and LDS dst
#define IPTR(p) ((int*)(void*)(p))

union H16 {
    v16h v;
    _Float16 h[16];
    unsigned u[8];
};

// 16-bit A-matrix (16x32) K-base per VGPR j and lane-half hi (ISA 7.12.2)
__device__ __forceinline__ int a_kbase(int j, int hi) {
    return (j < 4) ? (2 * j + 8 * hi) : (16 + 2 * (j - 4) + 8 * hi);
}
// 16-bit B-matrix (32x16) K-base per VGPR j and lane-half hi (same K order as A)
__device__ __forceinline__ int b_kbase(int j, int hi) {
    return (j < 4) ? (2 * j + 8 * hi) : (16 + 2 * (j - 4) + 8 * hi);
}

__device__ __forceinline__ void wait_async_zero() {
#if __has_builtin(__builtin_amdgcn_s_wait_asynccnt)
    __builtin_amdgcn_s_wait_asynccnt(0);
#else
    asm volatile("s_wait_asynccnt 0x0" ::: "memory");
#endif
}

__global__ __launch_bounds__(32)
void natten_wmma_kernel(const float* __restrict__ q,
                        const float* __restrict__ k,
                        const float* __restrict__ v,
                        float* __restrict__ out)
{
    __shared__ float    KsF[SLABP * DHEAD];   // K slab, f32, [pos][chan]
    __shared__ float    VsF[SLABPV * DHEAD];  // V slab, f32, zero-padded rows
    __shared__ _Float16 Pl[SLABPV * 16];      // attn probs, [slab][query]

    const int lane   = threadIdx.x & 31;
    const int lane16 = lane & 15;
    const int hi     = lane >> 4;

    int bid  = blockIdx.x;
    const int tile = bid % 144;  bid /= 144;
    const int head = bid % HEADS; bid /= HEADS;
    const int b    = bid;
    const int Y0 = (tile / 12) * TQ;
    const int X0 = (tile % 12) * TQ;
    const int c0 = head * DHEAD;

    const int plane = IMH * IMW; // 2304
    const float* qb = q + (size_t)(b * 256 + c0) * plane;
    const float* kb = k + (size_t)(b * 256 + c0) * plane;
    const float* vb = v + (size_t)(b * 256 + c0) * plane;

    // ---- stage K slab (112 rows) and V slab (128 rows) into LDS (f32) ----
    // Valid elements go straight global->LDS via the async data mover
    // (ASYNCcnt path); OOB/pad slots are zero-filled with ds stores.
    for (int e = lane; e < SLABP * DHEAD; e += 32) {
        const int p = e % SLABP, c = e / SLABP;
        const int gy = Y0 - RAD + p / SLAB;
        const int gx = X0 - RAD + p % SLAB;
        const bool valid = (p < SLAB2) &&
                           ((unsigned)gy < (unsigned)IMH) &&
                           ((unsigned)gx < (unsigned)IMW);
#if HAVE_ASYNC_LDS
        if (valid)
            __builtin_amdgcn_global_load_async_to_lds_b32(
                IPTR(kb + c * plane + gy * IMW + gx),
                IPTR(&KsF[p * DHEAD + c]), 0, 0);
        else
            KsF[p * DHEAD + c] = 0.f;
#else
        KsF[p * DHEAD + c] = valid ? kb[c * plane + gy * IMW + gx] : 0.f;
#endif
    }
    for (int e = lane; e < SLABPV * DHEAD; e += 32) {
        const int p = e % SLABPV, c = e / SLABPV;
        const int gy = Y0 - RAD + p / SLAB;
        const int gx = X0 - RAD + p % SLAB;
        const bool valid = (p < SLAB2) &&
                           ((unsigned)gy < (unsigned)IMH) &&
                           ((unsigned)gx < (unsigned)IMW);
#if HAVE_ASYNC_LDS
        if (valid)
            __builtin_amdgcn_global_load_async_to_lds_b32(
                IPTR(vb + c * plane + gy * IMW + gx),
                IPTR(&VsF[p * DHEAD + c]), 0, 0);
        else
            VsF[p * DHEAD + c] = 0.f;
#else
        VsF[p * DHEAD + c] = valid ? vb[c * plane + gy * IMW + gx] : 0.f;
#endif
    }

    // ---- load Q tile as WMMA A operand (scale folded in) ----
    const float scale = 0.17677669529663687f; // 1/sqrt(32)
    const int qy = lane16 >> 2, qx = lane16 & 3;
    const float* qpix = qb + (Y0 + qy) * IMW + (X0 + qx);
    H16 aQ;
#pragma unroll
    for (int j = 0; j < 8; ++j) {
        const int k0 = a_kbase(j, hi);
        aQ.h[2 * j]     = (_Float16)(qpix[k0 * plane] * scale);
        aQ.h[2 * j + 1] = (_Float16)(qpix[(k0 + 1) * plane] * scale);
    }

#if HAVE_ASYNC_LDS
    wait_async_zero();           // async copies landed in LDS
#endif
    __syncthreads();

    // ---- scores: S[16 x 112] = Q * K_slab^T, 7 WMMAs, K-dim = d = 32 ----
    v8f s[NT];
#pragma unroll
    for (int t = 0; t < NT; ++t) {
        const int n = t * 16 + lane16;   // slab position (B-matrix column)
        H16 bK;
#pragma unroll
        for (int j = 0; j < 8; ++j) {
            const int kbx = b_kbase(j, hi);     // contiguous f32 pair
            bK.h[2 * j]     = (_Float16)KsF[n * DHEAD + kbx];
            bK.h[2 * j + 1] = (_Float16)KsF[n * DHEAD + kbx + 1];
        }
        v8f cz = {};
        s[t] = __builtin_amdgcn_wmma_f32_16x16x32_f16(
                   false, aQ.v, false, bK.v, (short)0, cz, false, false);
    }

    // ---- mask out-of-window / out-of-image / pad columns ----
#pragma unroll
    for (int t = 0; t < NT; ++t) {
        const int ng = t * 16 + lane16;
        const int sy = ng / SLAB, sx = ng % SLAB;
        const int gy = Y0 - RAD + sy, gx = X0 - RAD + sx;
        const bool posok = (ng < SLAB2) &&
                           ((unsigned)gy < (unsigned)IMH) &&
                           ((unsigned)gx < (unsigned)IMW);
#pragma unroll
        for (int j = 0; j < 8; ++j) {
            const int mm  = j + 8 * hi;        // query row of this C element
            const int dy  = sy - RAD - (mm >> 2);
            const int dx  = sx - RAD - (mm & 3);
            const bool ok = posok && dy >= -RAD && dy <= RAD &&
                                     dx >= -RAD && dx <= RAD;
            if (!ok) s[t][j] = -1e30f;
        }
    }

    // ---- softmax per query row (row = fixed VGPR j within a 16-lane half) ----
#pragma unroll
    for (int j = 0; j < 8; ++j) {
        float mx = s[0][j];
#pragma unroll
        for (int t = 1; t < NT; ++t) mx = fmaxf(mx, s[t][j]);
#pragma unroll
        for (int off = 1; off <= 8; off <<= 1)
            mx = fmaxf(mx, __shfl_xor(mx, off, 32));
        float sm = 0.f;
#pragma unroll
        for (int t = 0; t < NT; ++t) {
            const float p = __expf(s[t][j] - mx);
            s[t][j] = p;
            sm += p;
        }
#pragma unroll
        for (int off = 1; off <= 8; off <<= 1)
            sm += __shfl_xor(sm, off, 32);
        const float inv = 1.0f / sm;
#pragma unroll
        for (int t = 0; t < NT; ++t)
            Pl[(t * 16 + lane16) * 16 + (j + 8 * hi)] =
                (_Float16)(s[t][j] * inv);
    }
    // zero the pad rows 112..127 of P
    for (int r = lane; r < (SLABPV - SLABP) * 16; r += 32)
        Pl[SLABP * 16 + r] = (_Float16)0.f;

    __syncthreads();

    // ---- output: O[16 x 32] = P[16 x 128] * V_slab[128 x 32], 8 WMMAs ----
    v8f o0 = {}, o1 = {};
#pragma unroll
    for (int kk = 0; kk < 4; ++kk) {
        H16 aP;
#pragma unroll
        for (int j = 0; j < 8; ++j) {
            const int k0 = kk * 32 + a_kbase(j, hi);
            aP.h[2 * j]     = Pl[k0 * 16 + lane16];
            aP.h[2 * j + 1] = Pl[(k0 + 1) * 16 + lane16];
        }
        H16 bV0, bV1;
#pragma unroll
        for (int j = 0; j < 8; ++j) {
            const int kbx = kk * 32 + b_kbase(j, hi);
            bV0.h[2 * j]     = (_Float16)VsF[kbx * DHEAD + lane16];
            bV0.h[2 * j + 1] = (_Float16)VsF[(kbx + 1) * DHEAD + lane16];
            bV1.h[2 * j]     = (_Float16)VsF[kbx * DHEAD + 16 + lane16];
            bV1.h[2 * j + 1] = (_Float16)VsF[(kbx + 1) * DHEAD + 16 + lane16];
        }
        o0 = __builtin_amdgcn_wmma_f32_16x16x32_f16(
                 false, aP.v, false, bV0.v, (short)0, o0, false, false);
        o1 = __builtin_amdgcn_wmma_f32_16x16x32_f16(
                 false, aP.v, false, bV1.v, (short)0, o1, false, false);
    }

    // ---- write result (C layout: n = lane16 -> channel, m in VGPRs) ----
    float* ob = out + (size_t)(b * 256 + c0) * plane;
#pragma unroll
    for (int j = 0; j < 8; ++j) {
        const int mm = j + 8 * hi;
        const int oy = Y0 + (mm >> 2), ox = X0 + (mm & 3);
        ob[lane16 * plane + oy * IMW + ox]        = o0[j];
        ob[(16 + lane16) * plane + oy * IMW + ox] = o1[j];
    }
}

extern "C" void kernel_launch(void* const* d_in, const int* in_sizes, int n_in,
                              void* d_out, int out_size, void* d_ws, size_t ws_size,
                              hipStream_t stream) {
    const float* q = (const float*)d_in[0];
    const float* k = (const float*)d_in[1];
    const float* v = (const float*)d_in[2];
    float* out = (float*)d_out;
    (void)in_sizes; (void)n_in; (void)out_size; (void)d_ws; (void)ws_size;

    dim3 grid(BATCH * HEADS * 12 * 12);  // one wave per 4x4 query tile
    dim3 block(32);                      // wave32
    hipLaunchKernelGGL(natten_wmma_kernel, grid, block, 0, stream,
                       q, k, v, out);
}
